// SelfAttention_11501922419040
// MI455X (gfx1250) — matrix operand order repaired and checked by reading, not compiled, not run on hardware
//
#include <hip/hip_runtime.h>

// ---------------------------------------------------------------------------
// Self-attention for MI455X (gfx1250, wave32, WMMA + Tensor Data Mover).
//   B=4, S=2048, E=1024, D=1024
// All matmuls via v_wmma_f32_16x16x32_bf16 (bf16 operands, f32 accumulate).
// bf16 tiles are staged into LDS by the TDM (tensor_load_to_lds) when the
// toolchain exposes the builtin; otherwise by manual ds staging.
// ---------------------------------------------------------------------------

typedef __bf16 bf16;
typedef __attribute__((ext_vector_type(16))) __bf16 v16bf;
typedef __attribute__((ext_vector_type(8)))  __bf16 v8bf;
typedef __attribute__((ext_vector_type(8)))  float  v8f;

#define BATCH 4
#define SEQ   2048
#define EMB   1024
#define DH    1024
#define BS    (BATCH * SEQ)          // 8192 rows for the projections

// Block tile: 128x128 output, K-step 32. 256 threads = 8 waves in a
// 4(M) x 2(N) grid; each wave owns a 32x64 strip = eight 16x16 f32
// accumulators -> 8 WMMAs per 12 LDS fragment loads per K-step.
#define BM 128
#define BN 128
#define BK 32
#define LDT 40                        // padded LDS row stride (bf16 elems)
#define TILE_BYTES (BM * LDT * 2)     // one staged operand tile in LDS

#if defined(__gfx1250__) && __has_builtin(__builtin_amdgcn_tensor_load_to_lds)
#define HAVE_TDM 1
#warning CDNA5_PROBE: tensor_load_to_lds builtin FOUND - TDM staging enabled
#else
#define HAVE_TDM 0
#if defined(__gfx1250__)
#warning CDNA5_PROBE: tensor_load_to_lds builtin MISSING - manual ds staging
#endif
#endif

__device__ __forceinline__ v8f wmma_bf16(v16bf a, v16bf b, v8f c) {
  // D = A(16x32) * B(32x16) + C, f32 accumulate
  return __builtin_amdgcn_wmma_f32_16x16x32_bf16(
      false, a, false, b, (short)0, c, false, false);
}

// Pack two bf16 (K-adjacent pair) into one dword for u32-granular transpose.
__device__ __forceinline__ unsigned pack_bf16(bf16 lo, bf16 hi) {
  unsigned short l = __builtin_bit_cast(unsigned short, lo);
  unsigned short h = __builtin_bit_cast(unsigned short, hi);
  return ((unsigned)h << 16) | (unsigned)l;
}

// A fragment (16x32 bf16) from LDS, rows at m0..m0+15, row-major stride LDT.
// ISA layout: lanes 0-15 hold M=lane, K = {0..7, 16..23}; lanes 16-31 hold
// M=lane-16, K = {8..15, 24..31}.  -> two 16-byte ds loads per lane.
__device__ __forceinline__ v16bf load_a_frag(const bf16* lA, int m0) {
  int lane = threadIdx.x & 31;
  int half = lane >> 4;
  int r    = lane & 15;
  const bf16* row = lA + (m0 + r) * LDT;
  v8bf lo = *(const v8bf*)(row + half * 8);
  v8bf hi = *(const v8bf*)(row + 16 + half * 8);
  v16bf a;
#pragma unroll
  for (int i = 0; i < 8; ++i) { a[i] = lo[i]; a[8 + i] = hi[i]; }
  return a;
}

// B fragment (32x16 bf16) from LDS staged N-major (lB[n][k], stride LDT).
// ISA layout: lanes 0-15 hold N=lane, K = {0..7, 16..23}; lanes 16-31 hold
// N=lane-16, K = {8..15, 24..31}. -> two 16-byte ds loads per lane.
__device__ __forceinline__ v16bf load_b_frag(const bf16* lB, int n0) {
  int lane = threadIdx.x & 31;
  int n  = lane & 15;
  int kh = lane >> 4;
  const bf16* p = lB + (n0 + n) * LDT + kh * 8;
  v8bf lo = *(const v8bf*)(p);
  v8bf hi = *(const v8bf*)(p + 16);
  v16bf b;
#pragma unroll
  for (int i = 0; i < 8; ++i) { b[i] = lo[i]; b[8 + i] = hi[i]; }
  return b;
}

// ---------------------------------------------------------------------------
// TDM 2D tile load: tile_d0 (contiguous, elements) x tile_d1 (rows) of bf16
// from a [tensor_d1 x tensor_d0] row-major tensor into LDS at lds_byte_off.
// pad_enable inserts 16B after every 64B row -> LDS row stride = 80B = LDT.
// D# bitfields per CDNA5 ISA.
// Issue from a single wave; EXEC is ignored by TENSOR_* ops.
// ---------------------------------------------------------------------------
__device__ __forceinline__ void tdm_load_2d(
    unsigned lds_byte_off, const bf16* gptr,
    unsigned tensor_d0, unsigned tensor_d1, unsigned stride0,
    unsigned tile_d0, unsigned tile_d1) {
#if HAVE_TDM
  typedef unsigned u32x4 __attribute__((ext_vector_type(4)));
  typedef int      i32x4 __attribute__((ext_vector_type(4)));
  typedef int      i32x8 __attribute__((ext_vector_type(8)));
  unsigned long long ga = (unsigned long long)(__UINTPTR_TYPE__)gptr;
  u32x4 g0;
  g0[0] = 1u;                                  // count=1 valid descriptor
  g0[1] = lds_byte_off;                        // lds_addr (bytes)
  g0[2] = (unsigned)ga;                        // global_addr[31:0]
  g0[3] = (unsigned)((ga >> 32) & 0x1FFFFFFu)  // global_addr[56:32]
          | (2u << 30);                        // type=2 ("image")
  i32x8 g1;
  g1[0] = (1 << 16)      // data_size = 1 -> 2-byte elements
        | (1 << 20)      // pad_enable
        | (3 << 22)      // pad_interval code 3 -> every 16 DWORDs (64B row)
        | (3 << 25);     // pad_amount  code 3 -> 4 DWORDs (16B)
  g1[1] = (int)(tensor_d0 << 16);                        // tensor_dim0 lo
  g1[2] = (int)((tensor_d0 >> 16) | (tensor_d1 << 16));  // dim0 hi | dim1 lo
  g1[3] = (int)((tensor_d1 >> 16) | (tile_d0 << 16));    // dim1 hi | tile_dim0
  g1[4] = (int)tile_d1;                                  // tile_dim1 (tile_dim2=0)
  g1[5] = (int)stride0;                                  // tensor_dim0_stride lo
  g1[6] = 0;
  g1[7] = 0;
  i32x4 z4 = {0, 0, 0, 0};
#if __clang_major__ >= 23
  i32x8 z8 = {0, 0, 0, 0, 0, 0, 0, 0};
  __builtin_amdgcn_tensor_load_to_lds(g0, g1, z4, z4, z8, 0);
#else
  __builtin_amdgcn_tensor_load_to_lds(g0, g1, z4, z4, 0);
#endif
#else
  (void)lds_byte_off; (void)gptr; (void)tensor_d0; (void)tensor_d1;
  (void)stride0; (void)tile_d0; (void)tile_d1;
#endif
}

__device__ __forceinline__ void tdm_wait() {
#if HAVE_TDM
#if __has_builtin(__builtin_amdgcn_s_wait_tensorcnt)
  __builtin_amdgcn_s_wait_tensorcnt(0);
#else
  asm volatile("s_wait_tensorcnt 0x0" ::: "memory");
#endif
#endif
}

// ---------------------------------------------------------------------------
// Projection GEMM: C[M,N](bf16) = A[M,K](f32) * W[K,N](f32), f32->bf16 on the
// fly during LDS staging (TDM cannot convert, so staging is manual here).
// M=BS, K=EMB, N=DH (all tile-divisible).
// ---------------------------------------------------------------------------
__global__ __launch_bounds__(256) void gemm_proj(
    const float* __restrict__ A, const float* __restrict__ W,
    bf16* __restrict__ C) {
  __shared__ alignas(16) bf16 smem[2 * BM * LDT];
  bf16* lA = smem;
  bf16* lB = smem + BM * LDT;

  const int tid  = threadIdx.x;
  const int wave = tid >> 5;
  const int wm   = wave & 3;          // wave M index (0..3), 32 rows each
  const int wn   = wave >> 2;         // wave N index (0..1), 64 cols each
  const int bm   = blockIdx.y * BM;
  const int bn   = blockIdx.x * BN;

  v8f acc[2][4] = {};

  for (int k0 = 0; k0 < EMB; k0 += BK) {
    // --- stage A tile 128x32: coalesced f32 reads, bf16 LDS rows ---
    {
      int r  = tid >> 1;              // 0..127
      int cg = (tid & 1) * 16;        // 0 or 16
      const float* src = A + (size_t)(bm + r) * EMB + k0 + cg;
      bf16* dst = lA + r * LDT + cg;
#pragma unroll
      for (int u = 0; u < 4; ++u) {
        float4 x = *(const float4*)(src + 4 * u);
        dst[4 * u + 0] = (bf16)x.x; dst[4 * u + 1] = (bf16)x.y;
        dst[4 * u + 2] = (bf16)x.z; dst[4 * u + 3] = (bf16)x.w;
      }
      if (k0 + BK < EMB) __builtin_prefetch(src + BK, 0, 0);
    }
    // --- stage W tile 32x128 transposed into lB[n][k], u32 (K-pair) grain ---
    {
      int kp = tid >> 4;              // K-pair 0..15 -> rows 2kp, 2kp+1
      int ng = (tid & 15) * 8;        // 0..120
      const float* s0 = W + (size_t)(k0 + 2 * kp) * DH + bn + ng;
      const float* s1 = s0 + DH;
      float4 x0 = *(const float4*)(s0);
      float4 x1 = *(const float4*)(s0 + 4);
      float4 y0 = *(const float4*)(s1);
      float4 y1 = *(const float4*)(s1 + 4);
      float lo[8] = {x0.x, x0.y, x0.z, x0.w, x1.x, x1.y, x1.z, x1.w};
      float hi[8] = {y0.x, y0.y, y0.z, y0.w, y1.x, y1.y, y1.z, y1.w};
      unsigned* dst = (unsigned*)lB;  // u32 view, row stride LDT/2 = 20
#pragma unroll
      for (int i = 0; i < 8; ++i)
        dst[(ng + i) * (LDT / 2) + kp] = pack_bf16((bf16)lo[i], (bf16)hi[i]);
      if (k0 + BK < EMB) __builtin_prefetch(s0 + (size_t)BK * DH, 0, 0);
    }
    __syncthreads();
    v16bf a0 = load_a_frag(lA, wm * 32);
    v16bf a1 = load_a_frag(lA, wm * 32 + 16);
    v16bf bf[4];
#pragma unroll
    for (int j = 0; j < 4; ++j) bf[j] = load_b_frag(lB, wn * 64 + 16 * j);
#pragma unroll
    for (int j = 0; j < 4; ++j) acc[0][j] = wmma_bf16(a0, bf[j], acc[0][j]);
#pragma unroll
    for (int j = 0; j < 4; ++j) acc[1][j] = wmma_bf16(a1, bf[j], acc[1][j]);
    __syncthreads();
  }

  // C layout: VGPR i -> row half*8+i, col = lane&15 (lanes 16-31 -> rows 8..15)
  const int lane = tid & 31;
  const int col  = lane & 15;
  const int half = lane >> 4;
#pragma unroll
  for (int mi = 0; mi < 2; ++mi)
#pragma unroll
    for (int nj = 0; nj < 4; ++nj)
#pragma unroll
      for (int i = 0; i < 8; ++i) {
        size_t row = (size_t)(bm + wm * 32 + mi * 16 + half * 8 + i);
        C[row * DH + bn + wn * 64 + nj * 16 + col] = (bf16)acc[mi][nj][i];
      }
}

// ---------------------------------------------------------------------------
// Scores GEMM (NT, batched over z): Z[b,q,k] = scale * Qp[b,q,:].Kp[b,k,:],
// masked to -1e9 where mask==0.  M=N=SEQ, K=DH.  Tiles staged by TDM.
// ---------------------------------------------------------------------------
__global__ __launch_bounds__(256) void gemm_scores(
    const bf16* __restrict__ Qp, const bf16* __restrict__ Kp,
    const int* __restrict__ mask, float* __restrict__ Z) {
  __shared__ alignas(16) bf16 smem[2 * BM * LDT];
  bf16* lA = smem;
  bf16* lB = smem + BM * LDT;

  const int tid  = threadIdx.x;
  const int wave = tid >> 5;
  const int wm   = wave & 3;
  const int wn   = wave >> 2;
  const int b    = blockIdx.z;
  const int bm   = blockIdx.y * BM;
  const int bn   = blockIdx.x * BN;

  const bf16* Qb = Qp + (size_t)b * SEQ * DH;
  const bf16* Kb = Kp + (size_t)b * SEQ * DH;

  v8f acc[2][4] = {};

  for (int k0 = 0; k0 < DH; k0 += BK) {
#if HAVE_TDM
    if (wave == 0) {
      // A tile: 32(K) x 128(M) rows of Q;  B tile: 32(K) x 128(N) rows of K
      tdm_load_2d(0,          Qb + (size_t)bm * DH + k0, DH, SEQ, DH, BK, BM);
      tdm_load_2d(TILE_BYTES, Kb + (size_t)bn * DH + k0, DH, SEQ, DH, BK, BN);
      tdm_wait();
    }
#else
    {   // A tile: Q rows, contiguous bf16
      int r  = tid >> 1;
      int cg = (tid & 1) * 16;
      const bf16* src = Qb + (size_t)(bm + r) * DH + k0 + cg;
      *(v8bf*)(lA + r * LDT + cg)     = *(const v8bf*)(src);
      *(v8bf*)(lA + r * LDT + cg + 8) = *(const v8bf*)(src + 8);
      if (k0 + BK < DH) __builtin_prefetch(src + BK, 0, 0);
    }
    {   // B tile (NT): K rows land directly in N-major LDS layout
      int n  = tid >> 1;
      int cg = (tid & 1) * 16;
      const bf16* src = Kb + (size_t)(bn + n) * DH + k0 + cg;
      *(v8bf*)(lB + n * LDT + cg)     = *(const v8bf*)(src);
      *(v8bf*)(lB + n * LDT + cg + 8) = *(const v8bf*)(src + 8);
      if (k0 + BK < DH) __builtin_prefetch(src + BK, 0, 0);
    }
#endif
    __syncthreads();
    v16bf a0 = load_a_frag(lA, wm * 32);
    v16bf a1 = load_a_frag(lA, wm * 32 + 16);
    v16bf bf[4];
#pragma unroll
    for (int j = 0; j < 4; ++j) bf[j] = load_b_frag(lB, wn * 64 + 16 * j);
#pragma unroll
    for (int j = 0; j < 4; ++j) acc[0][j] = wmma_bf16(a0, bf[j], acc[0][j]);
#pragma unroll
    for (int j = 0; j < 4; ++j) acc[1][j] = wmma_bf16(a1, bf[j], acc[1][j]);
    __syncthreads();
  }

  const float scale = 0.03125f;     // 1/sqrt(1024)
  const int lane = tid & 31;
  const int col  = lane & 15;
  const int half = lane >> 4;
#pragma unroll
  for (int mi = 0; mi < 2; ++mi)
#pragma unroll
    for (int nj = 0; nj < 4; ++nj)
#pragma unroll
      for (int i = 0; i < 8; ++i) {
        size_t row  = (size_t)(bm + wm * 32 + mi * 16 + half * 8 + i);
        size_t base = ((size_t)b * SEQ + row) * SEQ;
        int c = bn + wn * 64 + nj * 16 + col;
        float v = acc[mi][nj][i] * scale;
        Z[base + c] = (mask[base + c] == 0) ? -1e9f : v;
      }
}

// ---------------------------------------------------------------------------
// Row softmax: f32 scores -> bf16 probabilities.  One 256-thread block per row.
// ---------------------------------------------------------------------------
__global__ __launch_bounds__(256) void softmax_rows(
    const float* __restrict__ Z, bf16* __restrict__ P) {
  const size_t row = blockIdx.x;
  const float* z = Z + row * SEQ;
  bf16* p = P + row * SEQ;
  __shared__ float red[256];
  const int tid = threadIdx.x;

  float lmax = -3.4e38f;
  for (int c = tid; c < SEQ; c += 256) lmax = fmaxf(lmax, z[c]);
  red[tid] = lmax;
  __syncthreads();
  for (int s = 128; s > 0; s >>= 1) {
    if (tid < s) red[tid] = fmaxf(red[tid], red[tid + s]);
    __syncthreads();
  }
  const float m = red[0];
  __syncthreads();

  float lsum = 0.0f;
  for (int c = tid; c < SEQ; c += 256) lsum += __expf(z[c] - m);
  red[tid] = lsum;
  __syncthreads();
  for (int s = 128; s > 0; s >>= 1) {
    if (tid < s) red[tid] += red[tid + s];
    __syncthreads();
  }
  const float inv = 1.0f / red[0];

  for (int c = tid; c < SEQ; c += 256)
    p[c] = (bf16)(__expf(z[c] - m) * inv);
}

// ---------------------------------------------------------------------------
// AV GEMM (NN, batched): out[b,q,d] = sum_k P[b,q,k] * Vp[b,k,d].
// M=SEQ, K=SEQ, N=DH.  A tile staged by TDM; B tile needs a transpose so it
// is staged manually at u32 (K-pair) granularity (TDM cannot transpose).
// ---------------------------------------------------------------------------
__global__ __launch_bounds__(256) void gemm_av(
    const bf16* __restrict__ P, const bf16* __restrict__ Vp,
    float* __restrict__ O) {
  __shared__ alignas(16) bf16 smem[2 * BM * LDT];
  bf16* lA = smem;
  bf16* lB = smem + BM * LDT;

  const int tid  = threadIdx.x;
  const int wave = tid >> 5;
  const int wm   = wave & 3;
  const int wn   = wave >> 2;
  const int b    = blockIdx.z;
  const int bm   = blockIdx.y * BM;
  const int bn   = blockIdx.x * BN;

  const bf16* Pb = P  + (size_t)b * SEQ * SEQ;
  const bf16* Vb = Vp + (size_t)b * SEQ * DH;

  v8f acc[2][4] = {};

  for (int k0 = 0; k0 < SEQ; k0 += BK) {
#if HAVE_TDM
    if (wave == 0) {
      tdm_load_2d(0, Pb + (size_t)bm * SEQ + k0, SEQ, SEQ, SEQ, BK, BM);
      tdm_wait();
    }
#else
    {   // A tile: prob rows, contiguous bf16
      int r  = tid >> 1;
      int cg = (tid & 1) * 16;
      const bf16* src = Pb + (size_t)(bm + r) * SEQ + k0 + cg;
      *(v8bf*)(lA + r * LDT + cg)     = *(const v8bf*)(src);
      *(v8bf*)(lA + r * LDT + cg + 8) = *(const v8bf*)(src + 8);
      if (k0 + BK < SEQ) __builtin_prefetch(src + BK, 0, 0);
    }
#endif
    {   // B tile 32x128 (NN): transpose V rows into N-major LDS, u32 grain
      int kp = tid >> 4;              // K-pair 0..15
      int ng = (tid & 15) * 8;        // 0..120
      const bf16* s0 = Vb + (size_t)(k0 + 2 * kp) * DH + bn + ng;
      const bf16* s1 = s0 + DH;
      v8bf x = *(const v8bf*)(s0);
      v8bf y = *(const v8bf*)(s1);
      unsigned* dst = (unsigned*)lB;  // u32 view, row stride LDT/2 = 20
#pragma unroll
      for (int i = 0; i < 8; ++i)
        dst[(ng + i) * (LDT / 2) + kp] = pack_bf16(x[i], y[i]);
      if (k0 + BK < SEQ) __builtin_prefetch(s0 + (size_t)BK * DH, 0, 0);
    }
    __syncthreads();
    v16bf a0 = load_a_frag(lA, wm * 32);
    v16bf a1 = load_a_frag(lA, wm * 32 + 16);
    v16bf bf[4];
#pragma unroll
    for (int j = 0; j < 4; ++j) bf[j] = load_b_frag(lB, wn * 64 + 16 * j);
#pragma unroll
    for (int j = 0; j < 4; ++j) acc[0][j] = wmma_bf16(a0, bf[j], acc[0][j]);
#pragma unroll
    for (int j = 0; j < 4; ++j) acc[1][j] = wmma_bf16(a1, bf[j], acc[1][j]);
    __syncthreads();
  }

  const int lane = tid & 31;
  const int col  = lane & 15;
  const int half = lane >> 4;
#pragma unroll
  for (int mi = 0; mi < 2; ++mi)
#pragma unroll
    for (int nj = 0; nj < 4; ++nj)
#pragma unroll
      for (int i = 0; i < 8; ++i) {
        size_t row  = (size_t)(bm + wm * 32 + mi * 16 + half * 8 + i);
        size_t base = ((size_t)b * SEQ + row) * DH;
        O[base + bn + wn * 64 + nj * 16 + col] = acc[mi][nj][i];
      }
}

// ---------------------------------------------------------------------------
extern "C" void kernel_launch(void* const* d_in, const int* in_sizes, int n_in,
                              void* d_out, int out_size, void* d_ws, size_t ws_size,
                              hipStream_t stream) {
  (void)in_sizes; (void)n_in; (void)out_size; (void)ws_size;

  const float* q    = (const float*)d_in[0];
  const float* k    = (const float*)d_in[1];
  const float* v    = (const float*)d_in[2];
  const int*   mask = (const int*)  d_in[3];
  const float* Wq   = (const float*)d_in[4];
  const float* Wk   = (const float*)d_in[5];
  const float* Wv   = (const float*)d_in[6];
  float* out = (float*)d_out;

  // Workspace layout:
  //   Qp/Kp/Vp : 3 * 8192*1024 bf16  = 48 MB
  //   Zs       : 4*2048*2048 f32     = 64 MB
  //   P        : 4*2048*2048 bf16    = 32 MB
  bf16*  Qp = (bf16*)d_ws;
  bf16*  Kp = Qp + (size_t)BS * DH;
  bf16*  Vp = Kp + (size_t)BS * DH;
  float* Zs = (float*)(Vp + (size_t)BS * DH);
  bf16*  P  = (bf16*)(Zs + (size_t)BATCH * SEQ * SEQ);

  dim3 blk(256);

  // 1) Projections: [8192x1024] x [1024x1024] -> bf16
  dim3 gproj(DH / BN, BS / BM);
  gemm_proj<<<gproj, blk, 0, stream>>>(q, Wq, Qp);
  gemm_proj<<<gproj, blk, 0, stream>>>(k, Wk, Kp);
  gemm_proj<<<gproj, blk, 0, stream>>>(v, Wv, Vp);

  // 2) Scores: Q K^T * scale, masked -> f32
  dim3 gsc(SEQ / BN, SEQ / BM, BATCH);
  gemm_scores<<<gsc, blk, 0, stream>>>(Qp, Kp, mask, Zs);

  // 3) Row softmax -> bf16 probs
  softmax_rows<<<BATCH * SEQ, blk, 0, stream>>>(Zs, P);

  // 4) Output: P V -> f32
  dim3 gav(DH / BN, SEQ / BM, BATCH);
  gemm_av<<<gav, blk, 0, stream>>>(P, Vp, out);
}
